// DalleBartDecoder_45492293599409
// MI455X (gfx1250) — matrix operand order repaired and checked by reading, not compiled, not run on hardware
//
#include <hip/hip_runtime.h>
#include <hip/hip_bf16.h>

// ---------------------------------------------------------------------------
// MI455X (gfx1250) DALL-E BART decoder step.
//  - Heavy GEMMs: v_wmma_f32_16x16x32_bf16, f32 weights packed to bf16 with
//    one v_perm_b32 per pair (truncation), f32 accumulate. A tile double-
//    buffered in LDS (1 barrier per K-step), all addressing strength-reduced.
//  - 402 MB KV-cache pass-through: Tensor Data Mover (tensor_load_to_lds /
//    tensor_store_from_lds + s_wait_tensorcnt), 64 KB tiles per 1-wave WG.
//  - Workload is HBM-bound (~1.7 GB @ 23.3 TB/s ~= 73 us floor).
// ---------------------------------------------------------------------------

#define LNUM   12
#define E      1024
#define HN     16
#define HD     64
#define GDIM   2730
#define VOC    16416
#define IMG_T  256
#define B2     16
#define ICNT   8
#define TENC   64
#define CONDF  10.0f
#define KTOP   50

typedef __attribute__((ext_vector_type(16))) __bf16        v16bf;
typedef __attribute__((ext_vector_type(8)))  float         v8f;
typedef __attribute__((ext_vector_type(8)))  unsigned int  v8u;
typedef __attribute__((ext_vector_type(4)))  unsigned int  v4u;
typedef __attribute__((ext_vector_type(8)))  int           v8i;
typedef __attribute__((ext_vector_type(4)))  int           v4i;

// pack two f32 -> one dword of two bf16 (truncation), single v_perm_b32
__device__ __forceinline__ unsigned packbf(float fhi, float flo) {
  unsigned uh = __builtin_bit_cast(unsigned, fhi);
  unsigned ul = __builtin_bit_cast(unsigned, flo);
  return __builtin_amdgcn_perm(uh, ul, 0x07060302u);
}

__device__ __forceinline__ unsigned fkey(float f) {
  unsigned u = __builtin_bit_cast(unsigned, f);
  return (u & 0x80000000u) ? ~u : (u | 0x80000000u);
}

// ---------------------------------------------------------------------------
// GEMM: C[M,N] = A[M,K] @ W[K,N] (f32 in/out, bf16 WMMA inner).
// 128 threads = 4 waves; wave w owns the 16x16 tile at
// (blockIdx.y*16, blockIdx.x*64 + w*16).  M must be a multiple of 16.
// A tile (16x32) is double-buffered in LDS as packed bf16 dwords
// (row stride 17 dwords: no bank conflicts), 1 barrier per K-step.
// Main loop guard-free; K%32 tail handled once with guards; out-of-range
// columns clamped on load, masked on store.
// ---------------------------------------------------------------------------
__launch_bounds__(128)
__global__ void gemm_wmma_bf16(const float* __restrict__ A, int lda,
                               const float* __restrict__ W, int ldw,
                               float* __restrict__ C, int ldc,
                               int M, int N, int K)
{
  __shared__ unsigned As32[2][16 * 17];
  const int tid   = threadIdx.x;
  const int lane  = tid & 31;
  const int wid   = tid >> 5;
  const int mtile = blockIdx.y * 16;
  const int ntile = blockIdx.x * 64 + wid * 16;
  const int mr    = lane & 15;
  const int hi    = lane >> 4;             // 0: lanes 0-15, 1: lanes 16-31
  const int n     = ntile + mr;
  const int nc    = (n < N) ? n : (N - 1); // clamped (loads safe, store masked)
  const int K32   = K & ~31;

  // A staging: thread covers rows m0 = tid>>4 and m0+8, dword pair d = tid&15
  const int m0 = tid >> 4;
  const int d  = tid & 15;
  const float* apA = A + (size_t)(mtile + m0) * lda + 2 * d;  // apA[k0] = A[m0][k0+2d]
  const float* apB = apA + (size_t)8 * lda;
  const int ldsoff = m0 * 17 + d;

  // B addressing, strength-reduced
  const float* wpk = W + (size_t)(hi * 16) * ldw + nc;
  const size_t wstep = (size_t)32 * ldw;

  v8f acc = {};

  // prologue: stage k0=0 into buffer 0 (straight-line: 2 b64 loads, 2 packs)
  if (K32 > 0) {
    float a0 = apA[0], a1 = apA[1];
    float b0 = apB[0], b1 = apB[1];
    As32[0][ldsoff]          = packbf(a1, a0);
    As32[0][ldsoff + 8 * 17] = packbf(b1, b0);
  }

  const float* apAn = apA + 32;
  const float* apBn = apB + 32;
  int buf = 0;
  for (int k0 = 0; k0 < K32; k0 += 32) {
    __syncthreads();
    const bool more = (k0 + 32 < K32);
    // issue next A-tile loads early: they overlap the LDS reads + WMMA below
    float na0 = 0.f, na1 = 0.f, nb0 = 0.f, nb1 = 0.f;
    if (more) {
      na0 = apAn[0]; na1 = apAn[1];
      nb0 = apBn[0]; nb1 = apBn[1];
      apAn += 32; apBn += 32;
    }
    // B fragment for current k0: col = n, rows k0 + hi*16 + j (16 loads)
    v8u bu;
    {
      const float* wp = wpk;
#pragma unroll
      for (int t = 0; t < 8; ++t) {
        float w0 = wp[0];
        float w1 = wp[(size_t)ldw];
        bu[t] = packbf(w1, w0);
        wp += (size_t)2 * ldw;
      }
      wpk += wstep;
    }
    // A fragment from current LDS buffer (contiguous dword reads)
    v8u au;
    {
      const unsigned* Ab = As32[buf];
      int row = mr * 17, kb2 = hi * 8;
#pragma unroll
      for (int t = 0; t < 4; ++t) {
        au[t]     = Ab[row + kb2 + t];        // K = hi*16 + 2t, +1
        au[4 + t] = Ab[row + 4 + kb2 + t];    // K = hi*16 + 8 + 2t, +1
      }
    }
    // stage next A tile into the other buffer (no race: one barrier per step)
    if (more) {
      As32[buf ^ 1][ldsoff]          = packbf(na1, na0);
      As32[buf ^ 1][ldsoff + 8 * 17] = packbf(nb1, nb0);
    }
    acc = __builtin_amdgcn_wmma_f32_16x16x32_bf16(
        false, __builtin_bit_cast(v16bf, au),
        false, __builtin_bit_cast(v16bf, bu),
        (short)0, acc, false, false);
    buf ^= 1;
  }

  if (K32 < K) {   // K % 32 tail (only for K=2730), guarded + zero-padded
    __syncthreads();   // all readers of the last main buffer are done
    {
      int k = K32 + 2 * d;
      float a0 = (k     < K) ? apA[K32]     : 0.f;
      float a1 = (k + 1 < K) ? apA[K32 + 1] : 0.f;
      float b0 = (k     < K) ? apB[K32]     : 0.f;
      float b1 = (k + 1 < K) ? apB[K32 + 1] : 0.f;
      As32[0][ldsoff]          = packbf(a1, a0);
      As32[0][ldsoff + 8 * 17] = packbf(b1, b0);
    }
    __syncthreads();
    v8u au;
    {
      const unsigned* Ab = As32[0];
      int row = mr * 17, kb2 = hi * 8;
#pragma unroll
      for (int t = 0; t < 4; ++t) {
        au[t]     = Ab[row + kb2 + t];
        au[4 + t] = Ab[row + 4 + kb2 + t];
      }
    }
    v8u bu;
    {
      int kbase = K32 + hi * 16;
#pragma unroll
      for (int t = 0; t < 8; ++t) {
        int ka = kbase + 2 * t;
        float w0 = (ka     < K) ? W[(size_t)ka * ldw + nc]       : 0.f;
        float w1 = (ka + 1 < K) ? W[(size_t)(ka + 1) * ldw + nc] : 0.f;
        bu[t] = packbf(w1, w0);
      }
    }
    acc = __builtin_amdgcn_wmma_f32_16x16x32_bf16(
        false, __builtin_bit_cast(v16bf, au),
        false, __builtin_bit_cast(v16bf, bu),
        (short)0, acc, false, false);
  }

  // C/D layout: VGPR r -> row = mtile + hi*8 + r, col = n  (M % 16 == 0)
  if (n < N) {
#pragma unroll
    for (int r = 0; r < 8; ++r)
      C[(size_t)(mtile + hi * 8 + r) * ldc + n] = acc[r];
  }
}

// ---------------------------------------------------------------------------
// TDM cache pass-through: 64 KB (16384 f32) per 1-wave workgroup, DMA via LDS.
// ---------------------------------------------------------------------------
#define TDM_ELEMS 16384

__device__ __forceinline__ v4u tdm_group0(unsigned long long gaddr) {
  v4u g;
  g[0] = 1u;                                      // count=1, user descriptor
  g[1] = 0u;                                      // lds_addr = base of WG LDS
  g[2] = (unsigned)(gaddr & 0xFFFFFFFFu);         // global_addr[31:0]
  g[3] = (unsigned)((gaddr >> 32) & 0x01FFFFFFu)  // global_addr[56:32]
         | (2u << 30);                            // type = 2 ("image")
  return g;
}
__device__ __forceinline__ v8i tdm_group1_1d(void) {
  const unsigned elems = TDM_ELEMS;
  v8i g;
  g[0] = (int)(2u << 16);                    // workgroup_mask=0, data_size=4B
  g[1] = (int)((elems & 0xFFFFu) << 16);     // tensor_dim0[15:0] @ bit48
  g[2] = (int)((elems >> 16) | (1u << 16)); // tensor_dim0[31:16] | tensor_dim1=1
  g[3] = (int)(elems << 16);                 // tile_dim0 @ bit112 (16384)
  g[4] = 1;                                  // tile_dim1 = 1
  g[5] = (int)elems;                         // tensor_dim0_stride[31:0]
  g[6] = 0;
  g[7] = 0;
  return g;
}

__global__ void tdm_copy_kernel(const float* __restrict__ src, float* __restrict__ dst) {
#if defined(__AMDGCN__) && __has_builtin(__builtin_amdgcn_tensor_load_to_lds)
  size_t off = (size_t)blockIdx.x * TDM_ELEMS;
  v4u g0l = tdm_group0((unsigned long long)(src + off));
  v4u g0s = tdm_group0((unsigned long long)(dst + off));
  v8i g1  = tdm_group1_1d();
  v4i z4  = {0, 0, 0, 0};
#if defined(__clang_major__) && (__clang_major__ >= 23)
  v8i z8 = {0, 0, 0, 0, 0, 0, 0, 0};
  __builtin_amdgcn_tensor_load_to_lds(g0l, g1, z4, z4, z8, 0);
  __builtin_amdgcn_s_wait_tensorcnt((short)0);
  __builtin_amdgcn_tensor_store_from_lds(g0s, g1, z4, z4, z8, 0);
  __builtin_amdgcn_s_wait_tensorcnt((short)0);
#else
  __builtin_amdgcn_tensor_load_to_lds(g0l, g1, z4, z4, 0);
  __builtin_amdgcn_s_wait_tensorcnt((short)0);
  __builtin_amdgcn_tensor_store_from_lds(g0s, g1, z4, z4, 0);
  __builtin_amdgcn_s_wait_tensorcnt((short)0);
#endif
#else
  size_t base = (size_t)blockIdx.x * TDM_ELEMS;
  for (int i = threadIdx.x; i < TDM_ELEMS; i += blockDim.x)
    dst[base + i] = src[base + i];
#endif
}

// ---------------------------------------------------------------------------
// Block reductions (blockDim power of two)
// ---------------------------------------------------------------------------
__device__ __forceinline__ float block_sum(float v, float* red) {
  int tid = threadIdx.x;
  red[tid] = v; __syncthreads();
  for (int s = blockDim.x >> 1; s > 0; s >>= 1) {
    if (tid < s) red[tid] += red[tid + s];
    __syncthreads();
  }
  float r = red[0]; __syncthreads();
  return r;
}
__device__ __forceinline__ float block_max(float v, float* red) {
  int tid = threadIdx.x;
  red[tid] = v; __syncthreads();
  for (int s = blockDim.x >> 1; s > 0; s >>= 1) {
    if (tid < s) red[tid] = fmaxf(red[tid], red[tid + s]);
    __syncthreads();
  }
  float r = red[0]; __syncthreads();
  return r;
}

// ---------------------------------------------------------------------------
// Elementwise / LN kernels
// ---------------------------------------------------------------------------
__global__ void add_kernel(float* __restrict__ x, const float* __restrict__ t, int n) {
  int i = blockIdx.x * blockDim.x + threadIdx.x;
  if (i < n) x[i] += t[i];
}

__global__ void embed_ln_kernel(const int* __restrict__ prev, const int* __restrict__ tok,
                                const float* __restrict__ et, const float* __restrict__ ep,
                                const float* __restrict__ g, const float* __restrict__ b,
                                float* __restrict__ x) {
  __shared__ float red[256];
  int row = blockIdx.x;
  int t  = prev[row % ICNT];
  int ti = tok[0];
  const float* e0 = et + (size_t)t  * E;
  const float* e1 = ep + (size_t)ti * E;
  float s = 0.f, s2 = 0.f;
  for (int i = threadIdx.x; i < E; i += 256) { float v = e0[i] + e1[i]; s += v; s2 += v * v; }
  s  = block_sum(s,  red);
  s2 = block_sum(s2, red);
  float m = s / E, var = s2 / E - m * m;
  float inv = rsqrtf(var + 1e-5f);
  for (int i = threadIdx.x; i < E; i += 256)
    x[row * E + i] = ((e0[i] + e1[i]) - m) * inv * g[i] + b[i];
}

__global__ void ln_kernel(const float* __restrict__ in, const float* __restrict__ g,
                          const float* __restrict__ b, float* __restrict__ out, int width) {
  __shared__ float red[256];
  int row = blockIdx.x;
  const float* p = in + (size_t)row * width;
  float s = 0.f, s2 = 0.f;
  for (int i = threadIdx.x; i < width; i += 256) { float v = p[i]; s += v; s2 += v * v; }
  s  = block_sum(s,  red);
  s2 = block_sum(s2, red);
  float m = s / width, var = s2 / width - m * m;
  float inv = rsqrtf(var + 1e-5f);
  for (int i = threadIdx.x; i < width; i += 256)
    out[(size_t)row * width + i] = (p[i] - m) * inv * g[i] + b[i];
}

// out = res + LN(h)*g + b   (safe when out aliases res: elementwise)
__global__ void add_ln_kernel(const float* __restrict__ res, const float* __restrict__ h,
                              const float* __restrict__ g, const float* __restrict__ b,
                              float* __restrict__ out, int width) {
  __shared__ float red[256];
  int row = blockIdx.x;
  const float* p = h + (size_t)row * width;
  float s = 0.f, s2 = 0.f;
  for (int i = threadIdx.x; i < width; i += 256) { float v = p[i]; s += v; s2 += v * v; }
  s  = block_sum(s,  red);
  s2 = block_sum(s2, red);
  float m = s / width, var = s2 / width - m * m;
  float inv = rsqrtf(var + 1e-5f);
  for (int i = threadIdx.x; i < width; i += 256)
    out[(size_t)row * width + i] =
        res[(size_t)row * width + i] + (p[i] - m) * inv * g[i] + b[i];
}

// z = LN(gelu(g0) * g1) over G, with glu_ln1 affine
__global__ void glu_act_ln_kernel(const float* __restrict__ g0, const float* __restrict__ g1,
                                  const float* __restrict__ gg, const float* __restrict__ gb,
                                  float* __restrict__ z, int width) {
  __shared__ float red[256];
  int row = blockIdx.x;
  float s = 0.f, s2 = 0.f;
  for (int i = threadIdx.x; i < width; i += 256) {
    float a = g0[(size_t)row * width + i];
    float u = 0.5f * a * (1.f + erff(a * 0.70710678118f)) * g1[(size_t)row * width + i];
    z[(size_t)row * width + i] = u;
    s += u; s2 += u * u;
  }
  s  = block_sum(s,  red);
  s2 = block_sum(s2, red);
  float m = s / width, var = s2 / width - m * m;
  float inv = rsqrtf(var + 1e-5f);
  for (int i = threadIdx.x; i < width; i += 256)
    z[(size_t)row * width + i] = (z[(size_t)row * width + i] - m) * inv * gg[i] + gb[i];
}

// ---------------------------------------------------------------------------
// KV cache write + attention kernels
// ---------------------------------------------------------------------------
__global__ void write_kv_kernel(const float* __restrict__ kb, const float* __restrict__ vb,
                                float* __restrict__ ast_out, const int* __restrict__ tok,
                                int layer) {
  int r = blockIdx.x;                   // 0..31 : 0-15 K rows, 16-31 V rows
  int ti = tok[0];
  const float* src = (r < B2) ? (kb + (size_t)r * E) : (vb + (size_t)(r - B2) * E);
  float* dst = ast_out + ((((size_t)layer * 2 * B2 + r) * IMG_T) + ti) * E;
  for (int i = threadIdx.x; i < E; i += blockDim.x) dst[i] = src[i];
}

// grid = B2*HN blocks, 64 threads; q pre-proj [B2,E], cache in ast_out
__global__ void self_attn_kernel(const float* __restrict__ q,
                                 const float* __restrict__ ast_out,
                                 const int* __restrict__ tok,
                                 float* __restrict__ out, int layer) {
  __shared__ float qsh[HD], sc[IMG_T], red[64];
  int b = blockIdx.x >> 4, h = blockIdx.x & 15;
  int c = threadIdx.x;
  int nv = tok[0] + 1;
  const float* K  = ast_out + ((size_t)layer * 2 * B2 + b) * IMG_T * E;
  const float* Vv = ast_out + ((size_t)layer * 2 * B2 + B2 + b) * IMG_T * E;
  qsh[c] = q[(size_t)b * E + h * HD + c] * 0.125f;   // 1/sqrt(64)
  __syncthreads();
  float lm = -1e30f;
  for (int p = c; p < nv; p += 64) {
    const float* kp = K + (size_t)p * E + h * HD;
    float dot = 0.f;
    for (int e = 0; e < HD; ++e) dot += qsh[e] * kp[e];
    sc[p] = dot;
    lm = fmaxf(lm, dot);
  }
  float mx = block_max(lm, red);
  float ls = 0.f;
  for (int p = c; p < nv; p += 64) { float ex = expf(sc[p] - mx); sc[p] = ex; ls += ex; }
  float sum = block_sum(ls, red);
  float o = 0.f;
  for (int p = 0; p < nv; ++p) o += sc[p] * Vv[(size_t)p * E + h * HD + c];
  out[(size_t)b * E + h * HD + c] = o / sum;
}

// grid = B2*HN blocks, 64 threads; Kc/Vc [B2,TENC,E]
__global__ void cross_attn_kernel(const float* __restrict__ q,
                                  const float* __restrict__ Kc,
                                  const float* __restrict__ Vc,
                                  const unsigned char* __restrict__ mask,
                                  float* __restrict__ out) {
  __shared__ float qsh[HD], w[TENC], red[64];
  int b = blockIdx.x >> 4, h = blockIdx.x & 15;
  int c = threadIdx.x;
  qsh[c] = q[(size_t)b * E + h * HD + c] * 0.125f;
  __syncthreads();
  int p = c;
  const float* kp = Kc + ((size_t)b * TENC + p) * E + h * HD;
  float dot = 0.f;
  for (int e = 0; e < HD; ++e) dot += qsh[e] * kp[e];
  if (!mask[b * TENC + p]) dot -= 1e12f;
  float mx = block_max(dot, red);
  float ex = expf(dot - mx);
  float sum = block_sum(ex, red);
  w[c] = ex / sum;
  __syncthreads();
  float o = 0.f;
  for (int pp = 0; pp < TENC; ++pp)
    o += w[pp] * Vc[((size_t)b * TENC + pp) * E + h * HD + c];
  out[(size_t)b * E + h * HD + c] = o;
}

// ---------------------------------------------------------------------------
// CFG mix + top-50 threshold (binary search in monotonic uint key space)
// ---------------------------------------------------------------------------
__global__ void cfg_topk_kernel(const float* __restrict__ logits,
                                float* __restrict__ probs, float* __restrict__ mix) {
  __shared__ float red[256];
  __shared__ int   ired[256];
  int b = blockIdx.x;                   // 0..7
  int tid = threadIdx.x;
  float* m = mix + (size_t)b * VOC;
  for (int v = tid; v < VOC; v += 256)
    m[v] = logits[(size_t)b * VOC + v] * (1.f - CONDF) +
           logits[(size_t)(ICNT + b) * VOC + v] * CONDF;
  __syncthreads();
  float lm = -1e30f;
  for (int v = tid; v < VOC; v += 256) lm = fmaxf(lm, m[v]);
  float maxf = block_max(lm, red);
  unsigned lo = 0u, hi = 0xFFFFFFFFu;
  while (lo < hi) {
    unsigned mid = (unsigned)(((unsigned long long)lo + (unsigned long long)hi + 1ull) >> 1);
    int c = 0;
    for (int v = tid; v < VOC; v += 256) c += (fkey(m[v]) >= mid) ? 1 : 0;
    ired[tid] = c; __syncthreads();
    for (int s = 128; s > 0; s >>= 1) {
      if (tid < s) ired[tid] += ired[tid + s];
      __syncthreads();
    }
    int total = ired[0]; __syncthreads();
    if (total >= KTOP) lo = mid; else hi = mid - 1;
  }
  for (int v = tid; v < VOC; v += 256) {
    float x = m[v];
    probs[(size_t)b * VOC + v] = (fkey(x) < lo) ? 0.f : expf(x - maxf);
  }
}

// ---------------------------------------------------------------------------
static inline void gemm(const float* A, int lda, const float* W, int ldw,
                        float* C, int ldc, int M, int N, int K, hipStream_t s) {
  dim3 g((N + 63) / 64, (M + 15) / 16);
  gemm_wmma_bf16<<<g, dim3(128), 0, s>>>(A, lda, W, ldw, C, ldc, M, N, K);
}

extern "C" void kernel_launch(void* const* d_in, const int* in_sizes, int n_in,
                              void* d_out, int out_size, void* d_ws, size_t ws_size,
                              hipStream_t stream) {
  (void)in_sizes; (void)n_in; (void)out_size; (void)ws_size;
  const unsigned char* attn_mask = (const unsigned char*)d_in[0];
  const float* enc      = (const float*)d_in[1];
  const float* ast      = (const float*)d_in[2];
  const int*   prev     = (const int*)d_in[3];
  const int*   tok      = (const int*)d_in[4];
  const float* et       = (const float*)d_in[5];
  const float* ep       = (const float*)d_in[6];
  const float* ln_emb_g = (const float*)d_in[7];
  const float* ln_emb_b = (const float*)d_in[8];
  const float* final_g  = (const float*)d_in[9];
  const float* final_b  = (const float*)d_in[10];
  const float* lm_head  = (const float*)d_in[11];
  const float* sa_q = (const float*)d_in[12];
  const float* sa_k = (const float*)d_in[13];
  const float* sa_v = (const float*)d_in[14];
  const float* sa_o = (const float*)d_in[15];
  const float* ca_q = (const float*)d_in[16];
  const float* ca_k = (const float*)d_in[17];
  const float* ca_v = (const float*)d_in[18];
  const float* ca_o = (const float*)d_in[19];
  const float* lns_g = (const float*)d_in[20];
  const float* lns_b = (const float*)d_in[21];
  const float* glu1g = (const float*)d_in[22];
  const float* glu1b = (const float*)d_in[23];
  const float* fc0   = (const float*)d_in[24];
  const float* fc1   = (const float*)d_in[25];
  const float* fc2   = (const float*)d_in[26];

  float* out     = (float*)d_out;
  float* probs   = out;
  float* ast_out = out + (size_t)ICNT * VOC;      // probs first, then new_ast

  float* ws  = (float*)d_ws;
  float* xb  = ws;                                 // [16,1024]
  float* hb  = xb  + (size_t)B2 * E;
  float* qb  = hb  + (size_t)B2 * E;
  float* kb  = qb  + (size_t)B2 * E;
  float* vb  = kb  + (size_t)B2 * E;
  float* ab  = vb  + (size_t)B2 * E;
  float* h2b = ab  + (size_t)B2 * E;
  float* g0b = h2b + (size_t)B2 * E;               // [16,2730]
  float* g1b = g0b + (size_t)B2 * GDIM;
  float* zb  = g1b + (size_t)B2 * GDIM;
  float* Kc  = zb  + (size_t)B2 * GDIM;            // [16,64,1024]
  float* Vc  = Kc  + (size_t)B2 * TENC * E;
  float* lg  = Vc  + (size_t)B2 * TENC * E;        // logits [16,V]
  float* mix = lg  + (size_t)B2 * VOC;             // [8,V]

  // 1) full attention_state -> output cache via TDM DMA (6144 x 64KB tiles)
  {
    long long total = (long long)LNUM * 2 * B2 * IMG_T * E;   // 100,663,296 f32
    int nchunks = (int)(total / TDM_ELEMS);                    // 6144
    tdm_copy_kernel<<<nchunks, 32, 65536, stream>>>(ast, ast_out);
  }

  // 2) embeddings + LN -> x
  embed_ln_kernel<<<B2, 256, 0, stream>>>(prev, tok, et, ep, ln_emb_g, ln_emb_b, xb);

  const size_t WS = (size_t)E * E;
  const size_t FS = (size_t)E * GDIM;
  for (int l = 0; l < LNUM; ++l) {
    const float* lgp = lns_g + (size_t)l * 5 * E;
    const float* lbp = lns_b + (size_t)l * 5 * E;

    // ---- cached self-attention ----
    ln_kernel<<<B2, 256, 0, stream>>>(xb, lgp + 0 * E, lbp + 0 * E, hb, E);
    gemm(hb, E, sa_k + l * WS, E, kb, E, B2, E, E, stream);
    gemm(hb, E, sa_v + l * WS, E, vb, E, B2, E, E, stream);
    gemm(hb, E, sa_q + l * WS, E, qb, E, B2, E, E, stream);
    write_kv_kernel<<<2 * B2, 256, 0, stream>>>(kb, vb, ast_out, tok, l);
    self_attn_kernel<<<B2 * HN, HD, 0, stream>>>(qb, ast_out, tok, ab, l);
    gemm(ab, E, sa_o + l * WS, E, h2b, E, B2, E, E, stream);
    add_ln_kernel<<<B2, 256, 0, stream>>>(xb, h2b, lgp + 1 * E, lbp + 1 * E, xb, E);

    // ---- cross-attention (K/V recomputed: the big WMMA GEMMs) ----
    ln_kernel<<<B2, 256, 0, stream>>>(xb, lgp + 2 * E, lbp + 2 * E, hb, E);
    gemm(hb, E, ca_q + l * WS, E, qb, E, B2, E, E, stream);
    gemm(enc, E, ca_k + l * WS, E, Kc, E, B2 * TENC, E, E, stream);
    gemm(enc, E, ca_v + l * WS, E, Vc, E, B2 * TENC, E, E, stream);
    cross_attn_kernel<<<B2 * HN, HD, 0, stream>>>(qb, Kc, Vc, attn_mask, ab);
    gemm(ab, E, ca_o + l * WS, E, h2b, E, B2, E, E, stream);
    add_ln_kernel<<<B2, 256, 0, stream>>>(xb, h2b, lgp + 3 * E, lbp + 3 * E, xb, E);

    // ---- GLU ----
    ln_kernel<<<B2, 256, 0, stream>>>(xb, lgp + 4 * E, lbp + 4 * E, hb, E);
    gemm(hb, E, fc0 + l * FS, GDIM, g0b, GDIM, B2, GDIM, E, stream);
    gemm(hb, E, fc1 + l * FS, GDIM, g1b, GDIM, B2, GDIM, E, stream);
    glu_act_ln_kernel<<<B2, 256, 0, stream>>>(g0b, g1b, glu1g + (size_t)l * GDIM,
                                              glu1b + (size_t)l * GDIM, zb, GDIM);
    gemm(zb, GDIM, fc2 + l * (size_t)GDIM * E, E, h2b, E, B2, E, GDIM, stream);
    add_kernel<<<(B2 * E + 255) / 256, 256, 0, stream>>>(xb, h2b, B2 * E);
  }

  // 3) final LN + lm_head + CFG/top-k
  ln_kernel<<<B2, 256, 0, stream>>>(xb, final_g, final_b, hb, E);
  gemm(hb, E, lm_head, VOC, lg, VOC, B2, VOC, E, stream);
  cfg_topk_kernel<<<ICNT, 256, 0, stream>>>(lg, probs, mix);
}
